// Decoder_76888504533420
// MI455X (gfx1250) — compile-verified, matrix operand order checked by reading, not run on hardware
//
#include <hip/hip_runtime.h>
#include <hip/hip_bf16.h>
#include <math.h>

#define BB   64
#define TKN  400
#define HH   512
#define HD2  1024      // 2H
#define EMBD 256
#define VV   50000
#define OOVN 50
#define VO   (VV + OOVN)

typedef __attribute__((ext_vector_type(2)))  float    v2f;
typedef __attribute__((ext_vector_type(8)))  float    v8f;
typedef __attribute__((ext_vector_type(16))) _Float16 v16h;

__device__ __forceinline__ float sigm(float x) { return 1.0f / (1.0f + __expf(-x)); }

__device__ __forceinline__ float blk_sum(float v, float* sd) {
  int t = threadIdx.x;
  sd[t] = v; __syncthreads();
  for (int s = blockDim.x >> 1; s > 0; s >>= 1) {
    if (t < s) sd[t] += sd[t + s];
    __syncthreads();
  }
  float r = sd[0]; __syncthreads();
  return r;
}

__device__ __forceinline__ float blk_max(float v, float* sd) {
  int t = threadIdx.x;
  sd[t] = v; __syncthreads();
  for (int s = blockDim.x >> 1; s > 0; s >>= 1) {
    if (t < s) sd[t] = fmaxf(sd[t], sd[t + s]);
    __syncthreads();
  }
  float r = sd[0]; __syncthreads();
  return r;
}

// ---------------- 1) x = [c_t_1 | embed(y)] @ Wx^T + bx ----------------
__global__ void k_x(const float* __restrict__ c_t_1, const int* __restrict__ y,
                    const float* __restrict__ emb, const float* __restrict__ Wx,
                    const float* __restrict__ bx, float* __restrict__ x) {
  int idx = blockIdx.x * blockDim.x + threadIdx.x;      // BB*EMBD
  int b = idx >> 8, e = idx & 255;
  const float* wr = Wx + (size_t)e * (2 * HD2 + EMBD);
  const float4* w4 = (const float4*)wr;
  const float4* c4 = (const float4*)(c_t_1 + (size_t)b * (2 * HD2));
  float acc = bx[e];
  for (int k = 0; k < (2 * HD2) / 4; ++k) {
    float4 w = w4[k], c = c4[k];
    acc += w.x * c.x + w.y * c.y + w.z * c.z + w.w * c.w;
  }
  const float4* w4b = (const float4*)(wr + 2 * HD2);
  const float4* e4  = (const float4*)(emb + (size_t)y[b] * EMBD);
  for (int k = 0; k < EMBD / 4; ++k) {
    float4 w = w4b[k], c = e4[k];
    acc += w.x * c.x + w.y * c.y + w.z * c.z + w.w * c.w;
  }
  x[idx] = acc;
}

// ---------------- 2) single-step LSTM ----------------
__global__ void k_lstm(const float* __restrict__ x, const float* __restrict__ h0,
                       const float* __restrict__ c0,
                       const float* __restrict__ W_ih, const float* __restrict__ W_hh,
                       const float* __restrict__ b_ih, const float* __restrict__ b_hh,
                       float* __restrict__ h_new, float* __restrict__ c_new) {
  int idx = blockIdx.x * blockDim.x + threadIdx.x;      // BB*HH
  int b = idx / HH, h = idx % HH;
  const float4* xr = (const float4*)(x  + (size_t)b * EMBD);
  const float4* hr = (const float4*)(h0 + (size_t)b * HH);
  float g4[4];
#pragma unroll
  for (int q = 0; q < 4; ++q) {
    int row = q * HH + h;
    float acc = b_ih[row] + b_hh[row];
    const float4* wi = (const float4*)(W_ih + (size_t)row * EMBD);
    for (int k = 0; k < EMBD / 4; ++k) {
      float4 w = wi[k], v = xr[k];
      acc += w.x * v.x + w.y * v.y + w.z * v.z + w.w * v.w;
    }
    const float4* wh = (const float4*)(W_hh + (size_t)row * HH);
    for (int k = 0; k < HH / 4; ++k) {
      float4 w = wh[k], v = hr[k];
      acc += w.x * v.x + w.y * v.y + w.z * v.z + w.w * v.w;
    }
    g4[q] = acc;
  }
  float ci = sigm(g4[0]), cf = sigm(g4[1]), cg = tanhf(g4[2]), co = sigm(g4[3]);
  float cn = cf * c0[idx] + ci * cg;
  c_new[idx] = cn;
  h_new[idx] = co * tanhf(cn);
}

// ---------------- 3) dec_fea = s_t_hat @ Wp^T + bp (shared by both attns) ----------
__global__ void k_decfea(const float* __restrict__ h_new, const float* __restrict__ c_new,
                         const float* __restrict__ Wp, const float* __restrict__ bp,
                         float* __restrict__ dec_fea) {
  int idx = blockIdx.x * blockDim.x + threadIdx.x;      // BB*HD2
  int b = idx / HD2, n = idx % HD2;
  const float* wr = Wp + (size_t)n * HD2;
  const float4* w0 = (const float4*)wr;
  const float4* h4 = (const float4*)(h_new + (size_t)b * HH);
  const float4* c4 = (const float4*)(c_new + (size_t)b * HH);
  float acc = bp[n];
  for (int k = 0; k < HH / 4; ++k) {
    float4 w = w0[k], v = h4[k];
    acc += w.x * v.x + w.y * v.y + w.z * v.z + w.w * v.w;
  }
  const float4* w1 = (const float4*)(wr + HH);
  for (int k = 0; k < HH / 4; ++k) {
    float4 w = w1[k], v = c4[k];
    acc += w.x * v.x + w.y * v.y + w.z * v.z + w.w * v.w;
  }
  dec_fea[idx] = acc;
}

// ---------------- 4) scores for both encoders (dominant streaming pass) ----------
__global__ void k_scores(const float* __restrict__ gfeat, const float* __restrict__ afeat,
                         const float* __restrict__ dec_fea, const float* __restrict__ v_w,
                         float* __restrict__ gsc, float* __restrict__ asc) {
  __shared__ float sd[256];
  int bt = blockIdx.x;                                   // BB*TKN
  int b = bt / TKN;
  const float* gf = gfeat + (size_t)bt * HD2;
  const float* af = afeat + (size_t)bt * HD2;
  const float* df = dec_fea + (size_t)b * HD2;
  float pg = 0.f, pa = 0.f;
  for (int n = threadIdx.x; n < HD2; n += blockDim.x) {
    float d = df[n], v = v_w[n];
    pg += v * tanhf(gf[n] + d);
    pa += v * tanhf(af[n] + d);
  }
  float sg = blk_sum(pg, sd);
  float sa = blk_sum(pa, sd);
  if (threadIdx.x == 0) { gsc[bt] = sg; asc[bt] = sa; }
}

// ---------------- 5) masked softmax + renorm, in place ----------------
__global__ void k_softmax(float* __restrict__ gattn, float* __restrict__ aattn,
                          const float* __restrict__ mask) {
  __shared__ float sd[256];
  int e = blockIdx.x / BB, b = blockIdx.x % BB;
  float* s = (e == 0 ? gattn : aattn) + (size_t)b * TKN;
  const float* m = mask + (size_t)b * TKN;
  float mx = -3.4e38f;
  for (int t = threadIdx.x; t < TKN; t += blockDim.x) mx = fmaxf(mx, s[t]);
  mx = blk_max(mx, sd);
  float sum = 0.f;
  for (int t = threadIdx.x; t < TKN; t += blockDim.x) {
    float w = __expf(s[t] - mx) * m[t];
    s[t] = w; sum += w;
  }
  sum = blk_sum(sum, sd);
  float inv = 1.0f / sum;
  for (int t = threadIdx.x; t < TKN; t += blockDim.x) s[t] *= inv;
}

__global__ void k_attn_dist(const float* __restrict__ gattn, const float* __restrict__ aattn,
                            float* __restrict__ o_attn) {
  __shared__ float sd[256];
  int b = blockIdx.x;
  float sum = 0.f;
  for (int t = threadIdx.x; t < TKN; t += blockDim.x)
    sum += gattn[(size_t)b * TKN + t] + aattn[(size_t)b * TKN + t];
  sum = blk_sum(sum, sd);
  float inv = 1.0f / sum;
  for (int t = threadIdx.x; t < TKN; t += blockDim.x)
    o_attn[(size_t)b * TKN + t] =
        (gattn[(size_t)b * TKN + t] + aattn[(size_t)b * TKN + t]) * inv;
}

// ---------------- 6) c_t = attn @ enc_outputs (both encoders) ----------------
__global__ void k_ct(const float* __restrict__ g_out, const float* __restrict__ a_out,
                     const float* __restrict__ gattn, const float* __restrict__ aattn,
                     float* __restrict__ o_ct) {
  int idx = blockIdx.x * blockDim.x + threadIdx.x;       // BB * 2*HD2
  int b = idx / (2 * HD2), n = idx % (2 * HD2);
  int col = n & (HD2 - 1);
  const float* enc = (n < HD2 ? g_out : a_out) + (size_t)b * TKN * HD2 + col;
  const float* at  = (n < HD2 ? gattn : aattn) + (size_t)b * TKN;
  float acc = 0.f;
  for (int t = 0; t < TKN; ++t) acc += at[t] * enc[(size_t)t * HD2];
  o_ct[idx] = acc;
}

// ---------------- 7) p_gen ----------------
__global__ void k_pgen(const float* __restrict__ ct, const float* __restrict__ hn,
                       const float* __restrict__ cn, const float* __restrict__ x,
                       const float* __restrict__ Wpg, const float* __restrict__ bpg,
                       float* __restrict__ o_pgen) {
  __shared__ float sd[256];
  int b = blockIdx.x;
  float acc = 0.f;
  for (int k = threadIdx.x; k < 6 * HH + EMBD; k += blockDim.x) {
    float v;
    if (k < 4 * HH)      v = ct[(size_t)b * 4 * HH + k];
    else if (k < 5 * HH) v = hn[(size_t)b * HH + (k - 4 * HH)];
    else if (k < 6 * HH) v = cn[(size_t)b * HH + (k - 5 * HH)];
    else                 v = x[(size_t)b * EMBD + (k - 6 * HH)];
    acc += v * Wpg[k];
  }
  acc = blk_sum(acc, sd);
  if (threadIdx.x == 0) o_pgen[b] = sigm(acc + bpg[0]);
}

// ---------------- 8) out = tanh([h_new | c_t] @ W1^T + b1) ----------------
__global__ void k_outvec(const float* __restrict__ hn, const float* __restrict__ ct,
                         const float* __restrict__ W1, const float* __restrict__ b1,
                         float* __restrict__ outv) {
  int idx = blockIdx.x * blockDim.x + threadIdx.x;       // BB*HH
  int b = idx / HH, n = idx % HH;
  const float* wr = W1 + (size_t)n * (5 * HH);
  const float4* w0 = (const float4*)wr;
  const float4* h4 = (const float4*)(hn + (size_t)b * HH);
  float acc = b1[n];
  for (int k = 0; k < HH / 4; ++k) {
    float4 w = w0[k], v = h4[k];
    acc += w.x * v.x + w.y * v.y + w.z * v.z + w.w * v.w;
  }
  const float4* w1 = (const float4*)(wr + HH);
  const float4* c4 = (const float4*)(ct + (size_t)b * 4 * HH);
  for (int k = 0; k < (4 * HH) / 4; ++k) {
    float4 w = w1[k], v = c4[k];
    acc += w.x * v.x + w.y * v.y + w.z * v.z + w.w * v.w;
  }
  outv[idx] = tanhf(acc);
}

// ---------------- 9) logits = out @ W2^T + b2  (WMMA, fp32 end-to-end) ----------
// One wave computes one 16(M)x16(N) tile. 4 waves/block cover M=64; grid = VV/16.
__global__ void k_logits_wmma(const float* __restrict__ A,   // [64,512] out_vec
                              const float* __restrict__ W2,  // [50000,512]
                              const float* __restrict__ b2,
                              float* __restrict__ logits) {  // [64,50000]
  const int lane = threadIdx.x & 31;
  const int wave = threadIdx.x >> 5;          // M tile 0..3
  const int half = lane >> 4;                 // K-pair selector per ISA layout
  const int li   = lane & 15;
  const int mb   = wave * 16;
  const int nb   = blockIdx.x * 16;
  const float* arow = A  + (size_t)(mb + li) * HH;   // A row m (lanes 0-15 / 16-31 share M)
  const float* brow = W2 + (size_t)(nb + li) * HH;   // B col n == W2 row n
  v8f c = {};
#if __has_builtin(__builtin_amdgcn_wmma_f32_16x16x4_f32)
#pragma unroll 4
  for (int k0 = 0; k0 < HH; k0 += 4) {
    int ks = k0 + 2 * half;                   // lanes 0-15: K=k0,k0+1 ; lanes 16-31: K=k0+2,k0+3
    v2f av = *(const v2f*)(arow + ks);
    v2f bv = *(const v2f*)(brow + ks);
    c = __builtin_amdgcn_wmma_f32_16x16x4_f32(false, av, false, bv, (short)0, c,
                                              false, false);
  }
#else
  // fallback: f16 inputs, f32 accumulate (codegen-confirmed builtin)
  for (int k0 = 0; k0 < HH; k0 += 32) {
    v16h av{}, bv{};
#pragma unroll
    for (int j = 0; j < 16; ++j) {
      int kk = ((j >> 3) << 4) + (((j >> 1) & 3) << 1) + (j & 1) + (half << 3);
      av[j] = (_Float16)arow[k0 + kk];
      bv[j] = (_Float16)brow[k0 + kk];
    }
    c = __builtin_amdgcn_wmma_f32_16x16x32_f16(false, av, false, bv, (short)0, c,
                                               false, false);
  }
#endif
  const int col = nb + li;
  const float bias = b2[col];
#pragma unroll
  for (int r = 0; r < 8; ++r) {
    int row = mb + r + 8 * half;              // C layout: VGPR r -> M=r (lanes 0-15) / r+8
    logits[(size_t)row * VV + col] = c[r] + bias;
  }
}

// ---------------- 10) vocab softmax * p_gen  -> final_dist[:, :V], + extra_zeros ---
__global__ void k_final_vocab(const float* __restrict__ logits, const float* __restrict__ p_gen,
                              const float* __restrict__ extra_zeros,
                              float* __restrict__ final_dist) {
  __shared__ float sd[256];
  int b = blockIdx.x;
  const float* lr = logits + (size_t)b * VV;
  float* fr = final_dist + (size_t)b * VO;
  float mx = -3.4e38f;
  for (int v = threadIdx.x; v < VV; v += blockDim.x) mx = fmaxf(mx, lr[v]);
  mx = blk_max(mx, sd);
  float sum = 0.f;
  for (int v = threadIdx.x; v < VV; v += blockDim.x) sum += __expf(lr[v] - mx);
  sum = blk_sum(sum, sd);
  float scale = p_gen[b] / sum;
  for (int v = threadIdx.x; v < VV; v += blockDim.x) fr[v] = __expf(lr[v] - mx) * scale;
  for (int j = threadIdx.x; j < OOVN; j += blockDim.x)
    fr[VV + j] = extra_zeros[(size_t)b * OOVN + j];
}

// ---------------- 11) scatter-add pointer distribution + coverage passthrough -----
__global__ void k_scatter(const float* __restrict__ attn_dist, const float* __restrict__ p_gen,
                          const int* __restrict__ ext_vocab, const float* __restrict__ cov_in,
                          float* __restrict__ final_dist, float* __restrict__ cov_out) {
  int i = blockIdx.x * blockDim.x + threadIdx.x;         // BB*TKN
  int b = i / TKN;
  cov_out[i] = cov_in[i];
  float w = (1.0f - p_gen[b]) * attn_dist[i];
  atomicAdd(&final_dist[(size_t)b * VO + ext_vocab[i]], w);
}

extern "C" void kernel_launch(void* const* d_in, const int* in_sizes, int n_in,
                              void* d_out, int out_size, void* d_ws, size_t ws_size,
                              hipStream_t stream) {
  (void)in_sizes; (void)n_in; (void)out_size; (void)ws_size;
  const int*   y_t_1   = (const int*)  d_in[0];
  const float* h0      = (const float*)d_in[1];
  const float* c0      = (const float*)d_in[2];
  const float* g_out   = (const float*)d_in[3];
  const float* a_out   = (const float*)d_in[4];
  const float* g_feat  = (const float*)d_in[5];
  const float* a_feat  = (const float*)d_in[6];
  const float* mask    = (const float*)d_in[7];
  const float* c_t_1   = (const float*)d_in[8];
  const float* ezeros  = (const float*)d_in[9];
  const int*   extv    = (const int*)  d_in[10];
  const float* cov     = (const float*)d_in[11];
  const float* emb     = (const float*)d_in[12];
  const float* Wx      = (const float*)d_in[13];
  const float* bx      = (const float*)d_in[14];
  const float* W_ih    = (const float*)d_in[15];
  const float* W_hh    = (const float*)d_in[16];
  const float* b_ih    = (const float*)d_in[17];
  const float* b_hh    = (const float*)d_in[18];
  const float* Wp      = (const float*)d_in[19];
  const float* bp      = (const float*)d_in[20];
  const float* v_w     = (const float*)d_in[21];
  const float* Wpg     = (const float*)d_in[22];
  const float* bpg     = (const float*)d_in[23];
  const float* W1      = (const float*)d_in[24];
  const float* b1      = (const float*)d_in[25];
  const float* W2      = (const float*)d_in[26];
  const float* b2      = (const float*)d_in[27];

  // d_out layout: final_dist | h_new | c_new | c_t | attn_dist | p_gen | coverage
  float* out     = (float*)d_out;
  float* o_final = out;
  float* o_hnew  = o_final + (size_t)BB * VO;
  float* o_cnew  = o_hnew + (size_t)BB * HH;
  float* o_ct    = o_cnew + (size_t)BB * HH;
  float* o_attn  = o_ct + (size_t)BB * 2 * HD2;
  float* o_pgen  = o_attn + (size_t)BB * TKN;
  float* o_cov   = o_pgen + (size_t)BB;

  // workspace layout
  float* w        = (float*)d_ws;
  float* ws_x     = w;                                 // BB*EMBD
  float* ws_dec   = ws_x     + (size_t)BB * EMBD;      // BB*HD2
  float* ws_gattn = ws_dec   + (size_t)BB * HD2;       // BB*TKN
  float* ws_aattn = ws_gattn + (size_t)BB * TKN;       // BB*TKN
  float* ws_outv  = ws_aattn + (size_t)BB * TKN;       // BB*HH
  float* ws_logit = ws_outv  + (size_t)BB * HH;        // BB*VV

  k_x      <<<(BB * EMBD) / 256, 256, 0, stream>>>(c_t_1, y_t_1, emb, Wx, bx, ws_x);
  k_lstm   <<<(BB * HH) / 256, 256, 0, stream>>>(ws_x, h0, c0, W_ih, W_hh, b_ih, b_hh,
                                                 o_hnew, o_cnew);
  k_decfea <<<(BB * HD2) / 256, 256, 0, stream>>>(o_hnew, o_cnew, Wp, bp, ws_dec);
  k_scores <<<BB * TKN, 256, 0, stream>>>(g_feat, a_feat, ws_dec, v_w, ws_gattn, ws_aattn);
  k_softmax<<<2 * BB, 256, 0, stream>>>(ws_gattn, ws_aattn, mask);
  k_attn_dist<<<BB, 256, 0, stream>>>(ws_gattn, ws_aattn, o_attn);
  k_ct     <<<(BB * 2 * HD2) / 256, 256, 0, stream>>>(g_out, a_out, ws_gattn, ws_aattn, o_ct);
  k_pgen   <<<BB, 256, 0, stream>>>(o_ct, o_hnew, o_cnew, ws_x, Wpg, bpg, o_pgen);
  k_outvec <<<(BB * HH) / 256, 256, 0, stream>>>(o_hnew, o_ct, W1, b1, ws_outv);
  k_logits_wmma<<<VV / 16, 128, 0, stream>>>(ws_outv, W2, b2, ws_logit);
  k_final_vocab<<<BB, 256, 0, stream>>>(ws_logit, o_pgen, ezeros, o_final);
  k_scatter<<<(BB * TKN) / 256, 256, 0, stream>>>(o_attn, o_pgen, extv, cov, o_final, o_cov);
}
